// Swin_34540126994817
// MI455X (gfx1250) — matrix operand order repaired and checked by reading, not compiled, not run on hardware
//
#include <hip/hip_runtime.h>
#include <hip/hip_bf16.h>
#include <math.h>

typedef __attribute__((ext_vector_type(16))) _Float16 v16h;
typedef __attribute__((ext_vector_type(8)))  float    v8f;

#define NPTS   16384
#define DIMC   96
#define HEADS  6
#define HDIM   16
#define TC     288      // 3*DIM
#define MLPH   384
#define NWIN   512
#define WIN    32

// ---------------------------------------------------------------- utilities
__global__ void init_mins_k(int* mins) {
    if (threadIdx.x < 3) mins[threadIdx.x] = 0x7F7FFFFF; // +FLT_MAX bits
}

__global__ void min_xyz_k(const float* __restrict__ xyz, int* __restrict__ mins, int n) {
    int t = blockIdx.x * blockDim.x + threadIdx.x;
    if (t < n) {
        #pragma unroll
        for (int c = 0; c < 3; ++c)
            atomicMin(&mins[c], __float_as_int(xyz[(size_t)t * 3 + c])); // xyz >= 0
    }
}

__global__ void f32_to_f16_k(const float* __restrict__ in, _Float16* __restrict__ out, int n) {
    int t = blockIdx.x * blockDim.x + threadIdx.x;
    if (t < n) out[t] = (_Float16)in[t];
}

// ---------------------------------------------------------------- layernorm (one wave per row of 96)
__global__ __launch_bounds__(256) void ln_k(const float* __restrict__ x,
                                            const float* __restrict__ w,
                                            const float* __restrict__ b,
                                            _Float16* __restrict__ y, int nrows) {
    int lane = threadIdx.x & 31;
    int row  = blockIdx.x * 8 + (threadIdx.x >> 5);
    if (row >= nrows) return;
    const float* xr = x + (size_t)row * DIMC;
    float v0 = xr[lane], v1 = xr[lane + 32], v2 = xr[lane + 64];
    float s  = v0 + v1 + v2;
    float sq = v0 * v0 + v1 * v1 + v2 * v2;
    #pragma unroll
    for (int off = 16; off; off >>= 1) {
        s  += __shfl_xor(s,  off, 32);
        sq += __shfl_xor(sq, off, 32);
    }
    float mu  = s * (1.0f / 96.0f);
    float var = sq * (1.0f / 96.0f) - mu * mu;
    float rs  = rsqrtf(var + 1e-5f);
    _Float16* yr = y + (size_t)row * DIMC;
    yr[lane]      = (_Float16)((v0 - mu) * rs * w[lane]      + b[lane]);
    yr[lane + 32] = (_Float16)((v1 - mu) * rs * w[lane + 32] + b[lane + 32]);
    yr[lane + 64] = (_Float16)((v2 - mu) * rs * w[lane + 64] + b[lane + 64]);
}

// ---------------------------------------------------------------- WMMA GEMM: C = A(f16,MxK) * W(f16,NcxK)^T + bias (+epilogue)
// one 16x16 output tile per wave, v_wmma_f32_16x16x32_f16 over K
// mode 0: qkv  -> outF f32, scale first 96 cols by 0.25
// mode 1: +res -> outF f32
// mode 2: GELU -> outH f16
__global__ __launch_bounds__(128) void gemm_wmma_k(const _Float16* __restrict__ A,
                                                   const _Float16* __restrict__ W,
                                                   const float* __restrict__ bias,
                                                   const float* __restrict__ res,
                                                   float* __restrict__ outF,
                                                   _Float16* __restrict__ outH,
                                                   int M, int Nc, int K, int mode) {
    int wave = threadIdx.x >> 5;
    int tile = blockIdx.x * 4 + wave;
    int tiles_n = Nc >> 4;
    int tm = (tile / tiles_n) << 4;
    int tn = (tile % tiles_n) << 4;

    int l    = threadIdx.x & 31;
    int half = l >> 4;
    int r    = l & 15;

    v8f c = {0.f, 0.f, 0.f, 0.f, 0.f, 0.f, 0.f, 0.f};
    const _Float16* arow = A + (size_t)(tm + r) * K;
    const _Float16* brow = W + (size_t)(tn + r) * K;

    for (int kk = 0; kk < K; kk += 32) {
        v16h af, bf;
        const _Float16* ap = arow + kk + half * 8;      // K = kk + 8*half + e      (e=0..7)
        #pragma unroll
        for (int e = 0; e < 8; ++e) af[e] = ap[e];
        const _Float16* ap2 = ap + 16;                  // K = kk + 16 + 8*half + e (e=8..15)
        #pragma unroll
        for (int e = 0; e < 8; ++e) af[8 + e] = ap2[e];
        const _Float16* bp = brow + kk + half * 8;      // K = kk + 8*half + e (e<8), kk + 16 + 8*half + (e-8) (e>=8)
        #pragma unroll
        for (int e = 0; e < 16; ++e) bf[e] = bp[(e & 7) + ((e >> 3) << 4)];
        c = __builtin_amdgcn_wmma_f32_16x16x32_f16(false, af, false, bf,
                                                   (short)0, c, false, false);
    }

    int n = tn + r;
    float bi = bias[n];
    #pragma unroll
    for (int t = 0; t < 8; ++t) {
        int m = tm + t + 8 * half;
        float v = c[t] + bi;
        size_t idx = (size_t)m * Nc + n;
        if (mode == 0) {
            if (n < 96) v *= 0.25f;                      // q * head_dim^-0.5
            outF[idx] = v;
        } else if (mode == 1) {
            outF[idx] = v + res[idx];
        } else {
            float g = 0.5f * v * (1.0f + erff(v * 0.70710678118654752f)); // exact GELU
            outH[idx] = (_Float16)g;
        }
    }
}

// ---------------------------------------------------------------- window attention: one block per 32-point window
// thread (h = tid/32, i = tid%32) owns one (query,head) row of the 32x32 window
__global__ __launch_bounds__(192) void attn_k(const float* __restrict__ qkvf,  // N x 288
                                              const float* __restrict__ xyz,
                                              const float* __restrict__ table, // 64*6*16*3 f32
                                              const int*   __restrict__ minbits,
                                              const float* __restrict__ shiftp,
                                              _Float16* __restrict__ outH) {   // N x 96
    __shared__ _Float16 sq[WIN * DIMC];
    __shared__ _Float16 sk[WIN * DIMC];
    __shared__ _Float16 sv[WIN * DIMC];
    __shared__ _Float16 stab[64 * HEADS * HDIM * 3];   // 18432 halves
    __shared__ float    sxq[WIN][3];

    int tid  = threadIdx.x;
    int base = blockIdx.x * WIN;

    for (int idx = tid; idx < WIN * TC; idx += 192) {
        int i = idx / TC, col = idx % TC;
        float v = qkvf[(size_t)(base + i) * TC + col];
        if      (col < 96)  sq[i * DIMC + col]         = (_Float16)v;
        else if (col < 192) sk[i * DIMC + (col - 96)]  = (_Float16)v;
        else                sv[i * DIMC + (col - 192)] = (_Float16)v;
    }
    for (int idx = tid; idx < 64 * HEADS * HDIM * 3; idx += 192)
        stab[idx] = (_Float16)table[idx];

    if (tid < WIN) {
        float shift = shiftp[0];
        #pragma unroll
        for (int c = 0; c < 3; ++c) {
            float mn = __int_as_float(minbits[c]);
            float v  = xyz[(size_t)(base + tid) * 3 + c] - mn + shift;
            float rm = fmodf(v, 4.0f);
            if (rm < 0.f) rm += 4.0f;                  // python % semantics
            sxq[tid][c] = floorf(rm * 4.0f);           // / 0.25
        }
    }
    __syncthreads();

    int h = tid / WIN;
    int i = tid % WIN;

    float q[HDIM];
    #pragma unroll
    for (int d = 0; d < HDIM; ++d) q[d] = (float)sq[i * DIMC + h * HDIM + d];
    float xi0 = sxq[i][0], xi1 = sxq[i][1], xi2 = sxq[i][2];

    float s[WIN];
    float mx = -3.0e38f;
    #pragma unroll 4
    for (int j = 0; j < WIN; ++j) {
        int r0 = (int)(xi0 - sxq[j][0]) + 15;
        int r1 = (int)(xi1 - sxq[j][1]) + 15;
        int r2 = (int)(xi2 - sxq[j][2]) + 15;
        const _Float16* t0 = &stab[(r0 * HEADS + h) * 48 + 0];
        const _Float16* t1 = &stab[(r1 * HEADS + h) * 48 + 1];
        const _Float16* t2 = &stab[(r2 * HEADS + h) * 48 + 2];
        const _Float16* kj = &sk[j * DIMC + h * HDIM];
        float acc = 0.f;
        #pragma unroll
        for (int d = 0; d < HDIM; ++d) {
            float t = (float)t0[d * 3] + (float)t1[d * 3] + (float)t2[d * 3];
            acc += q[d] * ((float)kj[d] + t);
        }
        s[j] = acc;
        mx = fmaxf(mx, acc);
    }

    float denom = 0.f;
    float o[HDIM];
    #pragma unroll
    for (int d = 0; d < HDIM; ++d) o[d] = 0.f;
    #pragma unroll 4
    for (int j = 0; j < WIN; ++j) {
        float e = expf(s[j] - mx);
        denom += e;
        const _Float16* vj = &sv[j * DIMC + h * HDIM];
        #pragma unroll
        for (int d = 0; d < HDIM; ++d) o[d] += e * (float)vj[d];
    }
    float inv = 1.0f / denom;
    _Float16* orow = outH + (size_t)(base + i) * DIMC + h * HDIM;
    #pragma unroll
    for (int d = 0; d < HDIM; ++d) orow[d] = (_Float16)(o[d] * inv);
}

// ---------------------------------------------------------------- host
extern "C" void kernel_launch(void* const* d_in, const int* in_sizes, int n_in,
                              void* d_out, int out_size, void* d_ws, size_t ws_size,
                              hipStream_t stream) {
    const float* feats   = (const float*)d_in[0];
    const float* xyz     = (const float*)d_in[1];
    const float* shiftp  = (const float*)d_in[6];
    const float* norm1_w = (const float*)d_in[7];
    const float* norm1_b = (const float*)d_in[8];
    const float* qkv_w   = (const float*)d_in[9];
    const float* qkv_b   = (const float*)d_in[10];
    const float* reltab  = (const float*)d_in[11];
    const float* proj_w  = (const float*)d_in[12];
    const float* proj_b  = (const float*)d_in[13];
    const float* norm2_w = (const float*)d_in[14];
    const float* norm2_b = (const float*)d_in[15];
    const float* fc1_w   = (const float*)d_in[16];
    const float* fc1_b   = (const float*)d_in[17];
    const float* fc2_w   = (const float*)d_in[18];
    const float* fc2_b   = (const float*)d_in[19];
    float* outF = (float*)d_out;

    // workspace carve-up (256B aligned)
    size_t off = 0;
    auto carve = [&](size_t bytes) { size_t r = off; off = (off + bytes + 255) & ~(size_t)255; return r; };
    char* ws = (char*)d_ws;
    int*      mins   = (int*)     (ws + carve(16));
    _Float16* wqkv16 = (_Float16*)(ws + carve((size_t)TC   * DIMC * 2));
    _Float16* wproj16= (_Float16*)(ws + carve((size_t)DIMC * DIMC * 2));
    _Float16* wfc116 = (_Float16*)(ws + carve((size_t)MLPH * DIMC * 2));
    _Float16* wfc216 = (_Float16*)(ws + carve((size_t)DIMC * MLPH * 2));
    _Float16* x16    = (_Float16*)(ws + carve((size_t)NPTS * DIMC * 2));
    float*    qkvf   = (float*)   (ws + carve((size_t)NPTS * TC   * 4));
    _Float16* attn16 = (_Float16*)(ws + carve((size_t)NPTS * DIMC * 2));
    float*    feats2 = (float*)   (ws + carve((size_t)NPTS * DIMC * 4));
    _Float16* h16    = (_Float16*)(ws + carve((size_t)NPTS * DIMC * 2));
    _Float16* g16    = (_Float16*)(ws + carve((size_t)NPTS * MLPH * 2));

    // 1) mins init + reduce
    init_mins_k<<<1, 32, 0, stream>>>(mins);
    min_xyz_k<<<NPTS / 256, 256, 0, stream>>>(xyz, mins, NPTS);

    // 2) weights f32 -> f16 (native out x in layout == WMMA B layout)
    f32_to_f16_k<<<(TC * DIMC + 255) / 256, 256, 0, stream>>>(qkv_w, wqkv16, TC * DIMC);
    f32_to_f16_k<<<(DIMC * DIMC + 255) / 256, 256, 0, stream>>>(proj_w, wproj16, DIMC * DIMC);
    f32_to_f16_k<<<(MLPH * DIMC + 255) / 256, 256, 0, stream>>>(fc1_w, wfc116, MLPH * DIMC);
    f32_to_f16_k<<<(DIMC * MLPH + 255) / 256, 256, 0, stream>>>(fc2_w, wfc216, DIMC * MLPH);

    // 3) LN1 -> x16
    ln_k<<<NPTS / 8, 256, 0, stream>>>(feats, norm1_w, norm1_b, x16, NPTS);

    // 4) QKV GEMM (WMMA), q scaled in epilogue
    gemm_wmma_k<<<(NPTS / 16) * (TC / 16) / 4, 128, 0, stream>>>(
        x16, wqkv16, qkv_b, nullptr, qkvf, nullptr, NPTS, TC, DIMC, 0);

    // 5) windowed attention with rel-pos bias
    attn_k<<<NWIN, 192, 0, stream>>>(qkvf, xyz, reltab, mins, shiftp, attn16);

    // 6) proj GEMM + residual -> feats2
    gemm_wmma_k<<<(NPTS / 16) * (DIMC / 16) / 4, 128, 0, stream>>>(
        attn16, wproj16, proj_b, feats, feats2, nullptr, NPTS, DIMC, DIMC, 1);

    // 7) LN2 -> h16
    ln_k<<<NPTS / 8, 256, 0, stream>>>(feats2, norm2_w, norm2_b, h16, NPTS);

    // 8) fc1 GEMM + exact GELU -> g16
    gemm_wmma_k<<<(NPTS / 16) * (MLPH / 16) / 4, 128, 0, stream>>>(
        h16, wfc116, fc1_b, nullptr, nullptr, g16, NPTS, MLPH, DIMC, 2);

    // 9) fc2 GEMM + residual -> d_out
    gemm_wmma_k<<<(NPTS / 16) * (DIMC / 16) / 4, 128, 0, stream>>>(
        g16, wfc216, fc2_b, feats2, outF, nullptr, NPTS, DIMC, MLPH, 1);
}
